// HypoNet_43765716746933
// MI455X (gfx1250) — matrix operand order repaired and checked by reading, not compiled, not run on hardware
//
#include <hip/hip_runtime.h>
#include <hip/hip_bf16.h>

typedef _Float16 half_t;
typedef __attribute__((ext_vector_type(16))) _Float16 v16h;
typedef __attribute__((ext_vector_type(8)))  float    v8f;

union FragU { v16h v; unsigned int u[8]; uint4 q[2]; };

// ---------------------------------------------------------------------------
// Prep: pw = w[:-1]*m, column-L2-normalized; stored TRANSPOSED [O][Kpad] f16.
// Bias (last row of w, not normalized) stored separately as f32.
// ---------------------------------------------------------------------------
__global__ void prep_params(const float* __restrict__ w,
                            const float* __restrict__ m,
                            half_t* __restrict__ pT,
                            float* __restrict__ bias,
                            int K, int O, int Opad, int Kpad) {
    const int b = blockIdx.x;
    const int o = blockIdx.y * 64 + threadIdx.x;
    if (o >= Opad) return;
    half_t* row = pT + (size_t)(b * Opad + o) * Kpad;
    if (o < O) {
        float acc = 0.f;
        for (int k = 0; k < K; ++k) {
            float p = w[k * O + o] * m[((size_t)b * K + k) * O + o];
            acc += p * p;
        }
        float scale = 1.0f / fmaxf(sqrtf(acc), 1e-12f);
        for (int k = 0; k < K; ++k)
            row[k] = (half_t)(w[k * O + o] * m[((size_t)b * K + k) * O + o] * scale);
        for (int k = K; k < Kpad; ++k) row[k] = (half_t)0.f;
        bias[b * Opad + o] = w[K * O + o];
    } else {
        for (int k = 0; k < Kpad; ++k) row[k] = (half_t)0.f;
        bias[b * Opad + o] = 0.f;
    }
}

// Whole-layer weight copy (contiguous transposed block) into LDS.
__device__ __forceinline__ void copy_weights(half_t* __restrict__ dst,
                                             const half_t* __restrict__ src,
                                             int nhalf, int tid) {
    const int n4 = nhalf >> 3;  // uint4 = 8 halves
    for (int i = tid; i < n4; i += 256)
        ((uint4*)dst)[i] = ((const uint4*)src)[i];
}

// A fragment in this code's contiguous k order (same as B): row = mi*16 + ln;
// lane-half 0 holds K 0..15, half 1 holds K 16..31 (contiguous).
// Two 16B chunks, both 16B-aligned -> ds_load_b128 x2.
__device__ __forceinline__ void load_afrag(FragU& A, const half_t* __restrict__ hA,
                                           int mi, int ln, int lhalf, int ks) {
    const half_t* arow = hA + (mi * 16 + ln) * 128 + ks * 32 + 16 * lhalf;
    A.q[0] = *(const uint4*)(arow);
    A.q[1] = *(const uint4*)(arow + 8);
}

// B fragment: col = ni*16 + ln; lanes 0-15 K=0..15, 16-31 K=16..31.
// 16 contiguous halves, 32B-aligned base -> ds_load_b128 x2.
__device__ __forceinline__ void load_bfrag(FragU& B, const half_t* __restrict__ brow) {
    B.q[0] = *(const uint4*)(brow);
    B.q[1] = *(const uint4*)(brow + 8);
}

// ---------------------------------------------------------------------------
// Main: 1 block = 256 pixels of one batch through all 5 layers.
// 8 waves; wave w owns row-tiles w and w+8 for the entire network, so each
// B fragment read from LDS feeds two WMMAs.
// ---------------------------------------------------------------------------
__global__ __launch_bounds__(256)
void hyponet_main(const float* __restrict__ coord,
                  const float* __restrict__ latent,
                  const half_t* __restrict__ pT0, const half_t* __restrict__ pT1,
                  const half_t* __restrict__ pT2, const half_t* __restrict__ pT3,
                  const half_t* __restrict__ pT4,
                  const float* __restrict__ b0, const float* __restrict__ b1,
                  const float* __restrict__ b2, const float* __restrict__ b3,
                  const float* __restrict__ b4,
                  float* __restrict__ out) {
    const int tid   = threadIdx.x;
    const int wave  = tid >> 5;
    const int lane  = tid & 31;
    const int lhalf = lane >> 4;
    const int ln    = lane & 15;

    const int b      = blockIdx.x >> 8;     // 256 tiles per batch
    const int tile   = blockIdx.x & 255;
    const int pstart = tile << 8;           // 256 pixels per block

    __shared__ __align__(16) half_t hA[256 * 128];   // activations, 64 KB
    __shared__ __align__(16) half_t sW[128 * 128];   // full-layer weights, 32 KB

    // ---- Stage 0: bilinear upsample of latent + coords -> hA[p][0..34] ----
    {
        const int p  = tid;
        const int ng = pstart + p;
        const int hh = ng >> 8, ww = ng & 255;
        const float sy = (hh + 0.5f) * 0.25f - 0.5f;
        const float sx = (ww + 0.5f) * 0.25f - 0.5f;
        const float y0f = floorf(sy), x0f = floorf(sx);
        const float fy = sy - y0f, fx = sx - x0f;
        const int y0 = (int)y0f, x0 = (int)x0f;
        const int y0c = min(max(y0, 0), 63), y1c = min(max(y0 + 1, 0), 63);
        const int x0c = min(max(x0, 0), 63), x1c = min(max(x0 + 1, 0), 63);
        const float* Lb  = latent + (size_t)b * (64 * 64 * 32);
        const float* r00 = Lb + (y0c * 64 + x0c) * 32;
        const float* r01 = Lb + (y0c * 64 + x1c) * 32;
        const float* r10 = Lb + (y1c * 64 + x0c) * 32;
        const float* r11 = Lb + (y1c * 64 + x1c) * 32;
        const float w00 = (1.f - fy) * (1.f - fx), w01 = (1.f - fy) * fx;
        const float w10 = fy * (1.f - fx),         w11 = fy * fx;
        half_t* dst = hA + p * 128;
#pragma unroll 8
        for (int c = 0; c < 32; ++c)
            dst[c] = (half_t)(w00 * r00[c] + w01 * r01[c] + w10 * r10[c] + w11 * r11[c]);
        const float* cp = coord + ((size_t)b * 65536 + ng) * 3;
        dst[32] = (half_t)cp[0];
        dst[33] = (half_t)cp[1];
        dst[34] = (half_t)cp[2];
        for (int c = 35; c < 64; ++c) dst[c] = (half_t)0.f;  // layer0 K padded to 64
    }

    const half_t* pT[4]  = {pT0, pT1, pT2, pT3};
    const float*  bs[4]  = {b0, b1, b2, b3};
    const int ksteps[4]  = {2, 4, 4, 4};
    const int kpad[4]    = {64, 128, 128, 128};
    const v8f zero = {0.f, 0.f, 0.f, 0.f, 0.f, 0.f, 0.f, 0.f};

    // ---- Hidden layers 0..3 (128 outputs, sin activation) ----
    for (int layer = 0; layer < 4; ++layer) {
        const int KP = kpad[layer];
        copy_weights(sW, pT[layer] + (size_t)b * 128 * KP, 128 * KP, tid);
        if (layer < 3)
            __builtin_prefetch(pT[layer + 1] + (size_t)b * 128 * kpad[layer + 1], 0, 0);
        __syncthreads();   // weights visible; also orders stage-0 writes

        v8f c[2][8];
#pragma unroll
        for (int mt = 0; mt < 2; ++mt)
#pragma unroll
            for (int ni = 0; ni < 8; ++ni) c[mt][ni] = zero;

        const int KS = ksteps[layer];
        for (int ks = 0; ks < KS; ++ks) {
            FragU A0, A1;
            load_afrag(A0, hA, wave,     ln, lhalf, ks);
            load_afrag(A1, hA, wave + 8, ln, lhalf, ks);
#pragma unroll
            for (int ni = 0; ni < 8; ++ni) {
                FragU Bf;
                load_bfrag(Bf, sW + (ni * 16 + ln) * KP + ks * 32 + 16 * lhalf);
                c[0][ni] = __builtin_amdgcn_wmma_f32_16x16x32_f16(
                    false, A0.v, false, Bf.v, (short)0, c[0][ni], false, false);
                c[1][ni] = __builtin_amdgcn_wmma_f32_16x16x32_f16(
                    false, A1.v, false, Bf.v, (short)0, c[1][ni], false, false);
            }
        }

        // Epilogue: +bias, sin, write back to this wave's own rows.
        const float* bb = bs[layer] + b * 128;
#pragma unroll
        for (int mt = 0; mt < 2; ++mt) {
            const int mi = wave + 8 * mt;
#pragma unroll
            for (int ni = 0; ni < 8; ++ni) {
                const int n = ni * 16 + ln;
                const float bv = bb[n];
#pragma unroll
                for (int r = 0; r < 8; ++r) {
                    const int mrow = mi * 16 + r + 8 * lhalf;
                    hA[mrow * 128 + n] = (half_t)__sinf(c[mt][ni][r] + bv);
                }
            }
        }
        __syncthreads();   // all waves done reading sW before next layer reloads
    }

    // ---- Output layer (2 outputs padded to 16, no activation) ----
    {
        copy_weights(sW, pT4 + (size_t)b * 16 * 128, 16 * 128, tid);
        __syncthreads();
        v8f c4[2] = {zero, zero};
        for (int ks = 0; ks < 4; ++ks) {
            FragU A0, A1;
            load_afrag(A0, hA, wave,     ln, lhalf, ks);
            load_afrag(A1, hA, wave + 8, ln, lhalf, ks);
            FragU Bf;
            load_bfrag(Bf, sW + ln * 128 + ks * 32 + 16 * lhalf);
            c4[0] = __builtin_amdgcn_wmma_f32_16x16x32_f16(
                false, A0.v, false, Bf.v, (short)0, c4[0], false, false);
            c4[1] = __builtin_amdgcn_wmma_f32_16x16x32_f16(
                false, A1.v, false, Bf.v, (short)0, c4[1], false, false);
        }
        if (ln < 2) {
            const float bv = b4[b * 16 + ln];
#pragma unroll
            for (int mt = 0; mt < 2; ++mt) {
                const int mi = wave + 8 * mt;
#pragma unroll
                for (int r = 0; r < 8; ++r) {
                    const int mrow = mi * 16 + r + 8 * lhalf;
                    const int ng = pstart + mrow;
                    out[(size_t)b * 131072 + (size_t)ng * 2 + ln] = c4[mt][r] + bv;
                }
            }
        }
    }
}

// ---------------------------------------------------------------------------
extern "C" void kernel_launch(void* const* d_in, const int* in_sizes, int n_in,
                              void* d_out, int out_size, void* d_ws, size_t ws_size,
                              hipStream_t stream) {
    const float* coord  = (const float*)d_in[0];
    const float* latent = (const float*)d_in[1];
    const float* w[5] = {(const float*)d_in[2], (const float*)d_in[3],
                         (const float*)d_in[4], (const float*)d_in[5],
                         (const float*)d_in[6]};
    const float* m[5] = {(const float*)d_in[7], (const float*)d_in[8],
                         (const float*)d_in[9], (const float*)d_in[10],
                         (const float*)d_in[11]};
    float* out = (float*)d_out;

    char* ws = (char*)d_ws;
    half_t* pT0 = (half_t*)(ws + 0);        // 4*128*64*2  = 65536
    half_t* pT1 = (half_t*)(ws + 65536);    // 4*128*128*2 = 131072
    half_t* pT2 = (half_t*)(ws + 196608);
    half_t* pT3 = (half_t*)(ws + 327680);
    half_t* pT4 = (half_t*)(ws + 458752);   // 4*16*128*2  = 16384
    float*  bb0 = (float*)(ws + 475136);    // 4*128*4 = 2048
    float*  bb1 = (float*)(ws + 477184);
    float*  bb2 = (float*)(ws + 479232);
    float*  bb3 = (float*)(ws + 481280);
    float*  bb4 = (float*)(ws + 483328);    // 4*16*4 = 256

    prep_params<<<dim3(4, 2), 64, 0, stream>>>(w[0], m[0], pT0, bb0, 35, 128, 128, 64);
    prep_params<<<dim3(4, 2), 64, 0, stream>>>(w[1], m[1], pT1, bb1, 128, 128, 128, 128);
    prep_params<<<dim3(4, 2), 64, 0, stream>>>(w[2], m[2], pT2, bb2, 128, 128, 128, 128);
    prep_params<<<dim3(4, 2), 64, 0, stream>>>(w[3], m[3], pT3, bb3, 128, 128, 128, 128);
    prep_params<<<dim3(4, 1), 64, 0, stream>>>(w[4], m[4], pT4, bb4, 128, 2, 16, 128);

    hyponet_main<<<1024, 256, 0, stream>>>(coord, latent,
                                           pT0, pT1, pT2, pT3, pT4,
                                           bb0, bb1, bb2, bb3, bb4, out);
}
